// DecoderWithAttention_5068061409366
// MI455X (gfx1250) — matrix operand order repaired and checked by reading, not compiled, not run on hardware
//
#include <hip/hip_runtime.h>
#include <hip/hip_bf16.h>

// ---------------------------------------------------------------------------
// Problem sizes (fixed by the reference)
// ---------------------------------------------------------------------------
#define BB   128
#define PP   196
#define ENC  2048
#define AA   512
#define HH   512
#define EE   512
#define VV   10000
#define TT   20

typedef __bf16 bf16;
typedef __attribute__((ext_vector_type(16))) __bf16 v16bf;
typedef __attribute__((ext_vector_type(8)))  __bf16 v8bf;
typedef __attribute__((ext_vector_type(8)))  float  v8f;

// ---------------------------------------------------------------------------
// Helpers
// ---------------------------------------------------------------------------
__device__ __forceinline__ bf16 f2bf(float x) {
    union { float f; unsigned u; } v; v.f = x;
    unsigned r = v.u + 0x7FFFu + ((v.u >> 16) & 1u);   // round-to-nearest-even
    unsigned short h = (unsigned short)(r >> 16);
    bf16 out;
    __builtin_memcpy(&out, &h, 2);
    return out;
}
__device__ __forceinline__ float bf2f(bf16 x) {
    unsigned short h;
    __builtin_memcpy(&h, &x, 2);
    union { unsigned u; float f; } v; v.u = ((unsigned)h) << 16;
    return v.f;
}
// unpack a dword holding two packed bf16 values (little-endian: lo = element 0)
__device__ __forceinline__ void bfpk2f(unsigned u, float& v0, float& v1) {
    union { unsigned u; float f; } a, b;
    a.u = u << 16;
    b.u = u & 0xFFFF0000u;
    v0 = a.f; v1 = b.f;
}
__device__ __forceinline__ float sigf(float x) { return 1.f / (1.f + __expf(-x)); }

__device__ __forceinline__ float waveReduceSum(float v) {
    #pragma unroll
    for (int off = 16; off > 0; off >>= 1) v += __shfl_xor(v, off, 32);
    return v;
}
__device__ __forceinline__ float waveReduceMax(float v) {
    #pragma unroll
    for (int off = 16; off > 0; off >>= 1) v = fmaxf(v, __shfl_xor(v, off, 32));
    return v;
}

// ---------------------------------------------------------------------------
// Generic WMMA GEMM:  C[M,N] = rowScale(row) * (A[M,K](f32) * W[N,K](bf16)^T + bias[N])
//   - 128 threads = 4 wave32s per workgroup
//   - workgroup tile: 16 rows x 256 cols (4 waves x 4 sixteen-wide tiles)
//   - A tile staged fp32->bf16 in LDS once per K-step, shared by all 4 waves
//   - K loop fully branchless (tail N tiles clamped for loads, stores guarded
//     in the run-once epilogue) -> no EXEC predication around v_wmma.
//   - requires M % 16 == 0, K % 32 == 0 (true for every call site here)
// ---------------------------------------------------------------------------
#define GEMM_NW   4
#define GEMM_TPW  4

__global__ __launch_bounds__(128)
void wmma_gemm_kernel(const float* __restrict__ A, int lda,
                      const bf16* __restrict__ Bw,          // [N x K] row-major (= B^T)
                      const float* __restrict__ bias,       // [N] or nullptr
                      void* __restrict__ Cout, long long ldc,
                      int M, int N, int K,
                      const float* __restrict__ rowScale,   // [M] or nullptr
                      int outBf16)
{
    __shared__ __align__(16) bf16 lAs[16][40];   // 16x32 tile, stride 40 (aligned, conflict-free)

    const int tid   = threadIdx.x;
    const int wave  = tid >> 5;
    const int lane  = tid & 31;
    const int row16 = lane & 15;
    const int half  = lane >> 4;
    const int mBase = blockIdx.x * 16;
    const int nTileBase = blockIdx.y * (GEMM_NW * GEMM_TPW) + wave * GEMM_TPW;
    const int nTiles = (N + 15) >> 4;

    v8f acc[GEMM_TPW];
    #pragma unroll
    for (int j = 0; j < GEMM_TPW; ++j)
        #pragma unroll
        for (int i = 0; i < 8; ++i) acc[j][i] = 0.f;

    // Per-tile B base pointers, clamped to a valid tile (branchless hot loop).
    const bf16* bBase[GEMM_TPW];
    #pragma unroll
    for (int j = 0; j < GEMM_TPW; ++j) {
        int tj = nTileBase + j;
        tj = (tj < nTiles - 1) ? tj : (nTiles - 1);
        bBase[j] = Bw + (size_t)(tj * 16 + row16) * K + half * 16;
    }

    const int arow = tid >> 3;           // 0..15
    const int acol = (tid & 7) * 4;      // 0,4,...,28
    const float* aSrc = A + (size_t)(mBase + arow) * lda + acol;

    for (int kk = 0; kk < K; kk += 32) {
        __syncthreads();
        // stage A (16x32 fp32) -> LDS bf16; prefetch next K-step of the
        // streaming fp32 operand (global_prefetch_b8 on gfx1250)
        float4 a4 = *(const float4*)(aSrc + kk);
        if (kk + 32 < K) __builtin_prefetch(aSrc + kk + 32, 0, 0);
        lAs[arow][acol + 0] = f2bf(a4.x);
        lAs[arow][acol + 1] = f2bf(a4.y);
        lAs[arow][acol + 2] = f2bf(a4.z);
        lAs[arow][acol + 3] = f2bf(a4.w);
        __syncthreads();

        // A fragment: lane row=lane&15; K = half*16+{0..15} (contiguous, matches B)
        const v8bf alo = *(const v8bf*)&lAs[row16][half * 16];
        const v8bf ahi = *(const v8bf*)&lAs[row16][half * 16 + 8];
        const v16bf afrag = __builtin_shufflevector(alo, ahi,
            0, 1, 2, 3, 4, 5, 6, 7, 8, 9, 10, 11, 12, 13, 14, 15);

        #pragma unroll
        for (int j = 0; j < GEMM_TPW; ++j) {
            const bf16* bp = bBase[j] + kk;
            const v8bf blo = *(const v8bf*)bp;
            const v8bf bhi = *(const v8bf*)(bp + 8);
            const v16bf bfrag = __builtin_shufflevector(blo, bhi,
                0, 1, 2, 3, 4, 5, 6, 7, 8, 9, 10, 11, 12, 13, 14, 15);

            acc[j] = __builtin_amdgcn_wmma_f32_16x16x32_bf16(
                false, afrag, false, bfrag, (short)0, acc[j], false, false);
        }
    }

    // Epilogue (runs once): C/D layout — lane col = lane&15, rows r + (lane>>4)*8
    #pragma unroll
    for (int j = 0; j < GEMM_TPW; ++j) {
        const int nTile = nTileBase + j;
        if (nTile * 16 >= N) continue;                 // guard stores only
        const int col  = nTile * 16 + row16;
        const float bc = bias ? bias[col] : 0.f;
        #pragma unroll
        for (int r = 0; r < 8; ++r) {
            const int grow = mBase + half * 8 + r;
            float v = acc[j][r] + bc;
            if (rowScale) v *= rowScale[grow];
            if (outBf16)
                ((bf16*)Cout)[(size_t)grow * ldc + col] = f2bf(v);
            else
                ((float*)Cout)[(size_t)grow * ldc + col] = v;
        }
    }
}

// ---------------------------------------------------------------------------
// Setup kernels
// ---------------------------------------------------------------------------
// vectorized fp32 -> bf16 conversion (n must be a multiple of 4; true for all users)
__global__ __launch_bounds__(256)
void cvt4_bf16_kernel(const float* __restrict__ src, bf16* __restrict__ dst, int n4) {
    int i = blockIdx.x * 256 + threadIdx.x;
    if (i >= n4) return;
    float4 a = ((const float4*)src)[i];
    union { bf16 h[4]; unsigned long long u; } o;
    o.h[0] = f2bf(a.x); o.h[1] = f2bf(a.y); o.h[2] = f2bf(a.z); o.h[3] = f2bf(a.w);
    ((unsigned long long*)dst)[i] = o.u;
}

// Fuse W_ih (2048x2560) and W_hh (2048x512) into one bf16 matrix [2048 x 3072]
__global__ __launch_bounds__(256)
void build_wcat_kernel(const float* __restrict__ Wih, const float* __restrict__ Whh,
                       bf16* __restrict__ out) {
    int i = blockIdx.x * 256 + threadIdx.x;
    const int total = 2048 * 3072;
    if (i >= total) return;
    int nrow = i / 3072, k = i % 3072;
    float v = (k < 2560) ? Wih[(size_t)nrow * 2560 + k]
                         : Whh[(size_t)nrow * 512 + (k - 2560)];
    out[i] = f2bf(v);
}

// b_gates = b_ih + b_hh ; maskT[t*B + b] = (t < lengths[b]-1)
__global__ __launch_bounds__(256)
void bias_mask_kernel(const float* __restrict__ bih, const float* __restrict__ bhh,
                      float* __restrict__ bg, const int* __restrict__ lengths,
                      float* __restrict__ maskT) {
    int i = blockIdx.x * 256 + threadIdx.x;
    if (i < 2048) bg[i] = bih[i] + bhh[i];
    if (i < TT * BB) {
        int t = i / BB, b = i % BB;
        maskT[i] = (t < lengths[b] - 1) ? 1.f : 0.f;
    }
}

// mean over P of encoder_out -> [B, ENC]
__global__ __launch_bounds__(256)
void mean_kernel(const float* __restrict__ enc, float* __restrict__ meanOut) {
    int idx = blockIdx.x * 256 + threadIdx.x;          // < B*ENC
    int b = idx >> 11, e = idx & (ENC - 1);
    const float* eb = enc + (size_t)b * PP * ENC + e;
    float s = 0.f;
    for (int p = 0; p < PP; ++p) s += eb[(size_t)p * ENC];
    meanOut[idx] = s * (1.f / (float)PP);
}

// ---------------------------------------------------------------------------
// Attention score + softmax:  e[b,p] = sum_a relu(att1[b,p,a]+att2[b,a])*w[a] + b
// one workgroup (8 waves) per batch row; att1 is bf16 (L2-resident, 25.7MB).
// Packed 2xbf16 dword loads halve VMEM ops in the reduction.
// ---------------------------------------------------------------------------
__global__ __launch_bounds__(256)
void escore_softmax_kernel(const bf16* __restrict__ att1, const float* __restrict__ att2,
                           const float* __restrict__ w_full, const float* __restrict__ b_full,
                           float* __restrict__ alpha) {
    const int b = blockIdx.x;
    const int tid = threadIdx.x, lane = tid & 31, wave = tid >> 5;
    __shared__ float s_att2[AA], s_w[AA], s_e[256], s_red[8];

    for (int i = tid; i < AA; i += 256) { s_att2[i] = att2[b * AA + i]; s_w[i] = w_full[i]; }
    __syncthreads();
    const float bf = b_full[0];

    for (int p = wave; p < PP; p += 8) {
        const bf16* rowp = att1 + ((size_t)b * PP + p) * AA;
        float sum = 0.f;
        for (int a = lane * 2; a < AA; a += 64) {
            unsigned u = *(const unsigned*)(rowp + a);
            float v0, v1;
            bfpk2f(u, v0, v1);
            v0 = fmaxf(v0 + s_att2[a],     0.f);
            v1 = fmaxf(v1 + s_att2[a + 1], 0.f);
            sum = fmaf(v0, s_w[a], sum);
            sum = fmaf(v1, s_w[a + 1], sum);
        }
        sum = waveReduceSum(sum);
        if (lane == 0) s_e[p] = sum + bf;
    }
    __syncthreads();

    // softmax over PP values
    float v = (tid < PP) ? s_e[tid] : -3.4e38f;
    float m = waveReduceMax(v);
    if (lane == 0) s_red[wave] = m;
    __syncthreads();
    if (tid == 0) { float mm = s_red[0]; for (int i = 1; i < 8; ++i) mm = fmaxf(mm, s_red[i]); s_red[0] = mm; }
    __syncthreads();
    const float gmax = s_red[0];
    float ex = (tid < PP) ? __expf(v - gmax) : 0.f;
    float s = waveReduceSum(ex);
    __syncthreads();
    if (lane == 0) s_red[wave] = s;
    __syncthreads();
    if (tid == 0) { float ss = 0.f; for (int i = 0; i < 8; ++i) ss += s_red[i]; s_red[0] = ss; }
    __syncthreads();
    if (tid < PP) alpha[b * PP + tid] = ex / s_red[0];
}

// context[b,e] = sum_p alpha[b,p] * enc[b,p,e]  — fp32 path (fallback)
__global__ __launch_bounds__(256)
void context_kernel(const float* __restrict__ enc, const float* __restrict__ alpha,
                    float* __restrict__ ctx) {
    const int b = blockIdx.x, tid = threadIdx.x;
    __shared__ float s_a[PP];
    if (tid < PP) s_a[tid] = alpha[b * PP + tid];
    __syncthreads();
    const float* eb = enc + (size_t)b * PP * ENC;
    for (int e = tid; e < ENC; e += 256) {
        float acc = 0.f;
        #pragma unroll 4
        for (int p = 0; p < PP; ++p)
            acc = fmaf(s_a[p], eb[(size_t)p * ENC + e], acc);
        ctx[b * ENC + e] = acc;
    }
}

// context — bf16-encoder path: halves the dominant per-step HBM stream
// (103 MB/step instead of 205 MB/step). Packed 2xbf16 loads, 2 cols/thread.
__global__ __launch_bounds__(256)
void context_bf16_kernel(const bf16* __restrict__ encb, const float* __restrict__ alpha,
                         float* __restrict__ ctx) {
    const int b = blockIdx.x, tid = threadIdx.x;
    __shared__ float s_a[PP];
    if (tid < PP) s_a[tid] = alpha[b * PP + tid];
    __syncthreads();
    const bf16* eb = encb + (size_t)b * PP * ENC;
    for (int e = tid * 2; e < ENC; e += 512) {
        float a0 = 0.f, a1 = 0.f;
        #pragma unroll 4
        for (int p = 0; p < PP; ++p) {
            unsigned u = *(const unsigned*)(eb + (size_t)p * ENC + e);
            float v0, v1;
            bfpk2f(u, v0, v1);
            a0 = fmaf(s_a[p], v0, a0);
            a1 = fmaf(s_a[p], v1, a1);
        }
        ctx[b * ENC + e]     = a0;
        ctx[b * ENC + e + 1] = a1;
    }
}

// x_cat[b] = [ emb_table[cap] (512) | sigmoid(gatepre)*ctx (2048) | h (512) ]
__global__ __launch_bounds__(256)
void assemble_kernel(const float* __restrict__ emb_table, const int* __restrict__ captions,
                     const float* __restrict__ gatepre, const float* __restrict__ ctx,
                     const float* __restrict__ h, float* __restrict__ xcat, int t) {
    const int b = blockIdx.x, tid = threadIdx.x;
    const int cap = captions[b * (TT + 1) + t];
    float* xo = xcat + (size_t)b * 3072;
    for (int j = tid; j < 3072; j += 256) {
        float v;
        if (j < 512)       v = emb_table[(size_t)cap * EE + j];
        else if (j < 2560) { int e = j - 512; v = sigf(gatepre[b * ENC + e]) * ctx[b * ENC + e]; }
        else               v = h[b * HH + (j - 2560)];
        xo[j] = v;
    }
}

// LSTM pointwise: c,h update with length mask; hnew kept for the vocab GEMM
__global__ __launch_bounds__(256)
void lstm_kernel(const float* __restrict__ gates, float* __restrict__ h, float* __restrict__ c,
                 float* __restrict__ hnew, const float* __restrict__ maskT, int t) {
    const int idx = blockIdx.x * 256 + threadIdx.x;    // < B*H
    const int b = idx >> 9, j = idx & (HH - 1);
    const float* g = gates + (size_t)b * 2048;
    float gi = g[j], gf = g[512 + j], gg = g[1024 + j], go = g[1536 + j];
    float cn = sigf(gf) * c[idx] + sigf(gi) * tanhf(gg);
    float hn = sigf(go) * tanhf(cn);
    hnew[idx] = hn;
    float m = maskT[t * BB + b];
    h[idx] = m * hn + (1.f - m) * h[idx];
    c[idx] = m * cn + (1.f - m) * c[idx];
}

// ---------------------------------------------------------------------------
// Host orchestration
// ---------------------------------------------------------------------------
extern "C" void kernel_launch(void* const* d_in, const int* in_sizes, int n_in,
                              void* d_out, int out_size, void* d_ws, size_t ws_size,
                              hipStream_t stream) {
    const float* enc       = (const float*)d_in[0];
    const int*   captions  = (const int*)  d_in[1];
    const int*   lengths   = (const int*)  d_in[2];
    const float* W_enc_att = (const float*)d_in[3];
    const float* b_enc_att = (const float*)d_in[4];
    const float* W_dec_att = (const float*)d_in[5];
    const float* b_dec_att = (const float*)d_in[6];
    const float* w_full    = (const float*)d_in[7];
    const float* b_full    = (const float*)d_in[8];
    const float* emb_table = (const float*)d_in[9];
    const float* W_ih      = (const float*)d_in[10];
    const float* b_ih      = (const float*)d_in[11];
    const float* W_hh      = (const float*)d_in[12];
    const float* b_hh      = (const float*)d_in[13];
    const float* W_init_h  = (const float*)d_in[14];
    const float* b_init_h  = (const float*)d_in[15];
    const float* W_init_c  = (const float*)d_in[16];
    const float* b_init_c  = (const float*)d_in[17];
    const float* W_fbeta   = (const float*)d_in[18];
    const float* b_fbeta   = (const float*)d_in[19];
    const float* W_fc      = (const float*)d_in[20];
    const float* b_fc      = (const float*)d_in[21];
    float* out = (float*)d_out;

    // ---- workspace carve (~64 MB base; + optional 103 MB bf16 encoder copy) ----
    char* wp = (char*)d_ws;
    auto carve = [&](size_t bytes) -> void* {
        void* r = (void*)wp;
        wp += (bytes + 255) & ~(size_t)255;
        return r;
    };
    bf16* att1_bf = (bf16*)carve((size_t)BB * PP * AA * 2);
    bf16* Wenc_bf = (bf16*)carve((size_t)AA * ENC * 2);
    bf16* Wdec_bf = (bf16*)carve((size_t)AA * HH * 2);
    bf16* Wfb_bf  = (bf16*)carve((size_t)ENC * HH * 2);
    bf16* Wcat_bf = (bf16*)carve((size_t)2048 * 3072 * 2);
    bf16* Wfc_bf  = (bf16*)carve((size_t)VV * HH * 2);
    bf16* Wih0_bf = (bf16*)carve((size_t)HH * ENC * 2);
    bf16* Wic0_bf = (bf16*)carve((size_t)HH * ENC * 2);
    float* b_gates = (float*)carve(2048 * 4);
    float* maskT   = (float*)carve((size_t)TT * BB * 4);
    float* meanBuf = (float*)carve((size_t)BB * ENC * 4);
    float* hBuf    = (float*)carve((size_t)BB * HH * 4);
    float* cBuf    = (float*)carve((size_t)BB * HH * 4);
    float* hnewBuf = (float*)carve((size_t)BB * HH * 4);
    float* att2Buf = (float*)carve((size_t)BB * AA * 4);
    float* alphaBuf= (float*)carve((size_t)BB * PP * 4);
    float* ctxBuf  = (float*)carve((size_t)BB * ENC * 4);
    float* gateBuf = (float*)carve((size_t)BB * ENC * 4);
    float* xcatBuf = (float*)carve((size_t)BB * 3072 * 4);
    float* gatesBuf= (float*)carve((size_t)BB * 2048 * 4);

    // bf16 encoder copy carved last; only used if the workspace can hold it.
    const size_t encElems = (size_t)BB * PP * ENC;
    const size_t usedSoFar = (size_t)(wp - (char*)d_ws);
    const bool useEncBf = (ws_size >= usedSoFar + encElems * 2 + 256);
    bf16* enc_bf = useEncBf ? (bf16*)carve(encElems * 2) : nullptr;

    auto cvt = [&](const float* s, bf16* d, size_t n) {
        int n4 = (int)(n / 4);
        cvt4_bf16_kernel<<<(n4 + 255) / 256, 256, 0, stream>>>(s, d, n4);
    };
    cvt(W_enc_att, Wenc_bf, (size_t)AA * ENC);
    cvt(W_dec_att, Wdec_bf, (size_t)AA * HH);
    cvt(W_fbeta,   Wfb_bf,  (size_t)ENC * HH);
    cvt(W_fc,      Wfc_bf,  (size_t)VV * HH);
    cvt(W_init_h,  Wih0_bf, (size_t)HH * ENC);
    cvt(W_init_c,  Wic0_bf, (size_t)HH * ENC);
    if (useEncBf) cvt(enc, enc_bf, encElems);
    build_wcat_kernel<<<(2048 * 3072 + 255) / 256, 256, 0, stream>>>(W_ih, W_hh, Wcat_bf);
    bias_mask_kernel<<<(TT * BB + 255) / 256 + 8, 256, 0, stream>>>(b_ih, b_hh, b_gates, lengths, maskT);
    mean_kernel<<<(BB * ENC) / 256, 256, 0, stream>>>(enc, meanBuf);

    auto gemm = [&](const float* A, int lda, const bf16* Bm, const float* bias,
                    void* C, long long ldc, int M, int N, int K,
                    const float* rs, int obf) {
        dim3 grid(M / 16, (N + 255) / 256);
        wmma_gemm_kernel<<<grid, 128, 0, stream>>>(A, lda, Bm, bias, C, ldc, M, N, K, rs, obf);
    };

    // h0, c0 from mean encoder state
    gemm(meanBuf, ENC, Wih0_bf, b_init_h, hBuf, HH, BB, HH, ENC, nullptr, 0);
    gemm(meanBuf, ENC, Wic0_bf, b_init_c, cBuf, HH, BB, HH, ENC, nullptr, 0);
    // att1 = enc @ W_enc_att^T + b  (stored bf16, L2-resident across all steps)
    gemm(enc, ENC, Wenc_bf, b_enc_att, att1_bf, AA, BB * PP, AA, ENC, nullptr, 1);

    for (int t = 0; t < TT; ++t) {
        gemm(hBuf, HH, Wdec_bf, b_dec_att, att2Buf, AA, BB, AA, HH, nullptr, 0);
        escore_softmax_kernel<<<BB, 256, 0, stream>>>(att1_bf, att2Buf, w_full, b_full, alphaBuf);
        if (useEncBf)
            context_bf16_kernel<<<BB, 256, 0, stream>>>(enc_bf, alphaBuf, ctxBuf);
        else
            context_kernel<<<BB, 256, 0, stream>>>(enc, alphaBuf, ctxBuf);
        gemm(hBuf, HH, Wfb_bf, b_fbeta, gateBuf, ENC, BB, ENC, HH, nullptr, 0);
        assemble_kernel<<<BB, 256, 0, stream>>>(emb_table, captions, gateBuf, ctxBuf, hBuf, xcatBuf, t);
        gemm(xcatBuf, 3072, Wcat_bf, b_gates, gatesBuf, 2048, BB, 2048, 3072, nullptr, 0);
        lstm_kernel<<<(BB * HH) / 256, 256, 0, stream>>>(gatesBuf, hBuf, cBuf, hnewBuf, maskT, t);
        // pred_t = (h_new @ W_fc^T + b_fc) * mask  -> d_out[b, t, :]
        gemm(hnewBuf, HH, Wfc_bf, b_fc, out + (size_t)t * VV, (long long)TT * VV,
             BB, VV, HH, maskT + t * BB, 0);
    }
}
